// DataReuploadingTorso_83769042141231
// MI455X (gfx1250) — matrix operand order repaired and checked by reading, not compiled, not run on hardware
//
#include <hip/hip_runtime.h>

// ---------------------------------------------------------------------------
// Quantum data-reuploading circuit, fused single kernel for MI455X (gfx1250).
//  - wave32: one wave simulates one batch item; 256 complex amps distributed
//    8 per lane (16 VGPRs). Qubit butterflies: bits 0-2 in-register, bits 3-7
//    via wave32 shuffles (ds path).
//  - Final <Z_q> = probs(16x256) @ Z(256x8) done with v_wmma_f32_16x16x32_f16
//    per workgroup (16 items/WG staged through padded LDS as f16).
//  - Z operand built with integer bit tricks (f16 ±1 = 0x3C00 ^ sign).
// ---------------------------------------------------------------------------

typedef __attribute__((ext_vector_type(16))) _Float16     v16h;
typedef __attribute__((ext_vector_type(8)))  float        v8f;
typedef __attribute__((ext_vector_type(8)))  unsigned int v8u;

#define NQ      8
#define DIM     256
#define NL      4
#define ITEMS   16          // items (waves) per workgroup
#define THREADS 512         // 16 waves * 32 lanes
#define LDSTR   264         // 256 + 8 halves padding (breaks LDS bank conflicts)

struct C2 { float x, y; };

__device__ __forceinline__ C2 cmadd2(C2 m0, C2 a0, C2 m1, C2 a1) {
    // m0*a0 + m1*a1 (complex)
    C2 r;
    r.x = m0.x * a0.x - m0.y * a0.y + m1.x * a1.x - m1.y * a1.y;
    r.y = m0.x * a0.y + m0.y * a0.x + m1.x * a1.y + m1.y * a1.x;
    return r;
}

// PennyLane Rot(phi, theta, omega) = RZ(omega) RY(theta) RZ(phi)
__device__ __forceinline__ void rot_mat(float phi, float th, float om, C2 m[4]) {
    float c, s, ca, sa, cb, sb;
    __sincosf(0.5f * th, &s, &c);           // native v_sin/v_cos path
    __sincosf(0.5f * (phi + om), &sa, &ca); // alpha
    __sincosf(0.5f * (phi - om), &sb, &cb); // beta
    m[0] = { ca * c, -sa * c };             // e^{-i a} c
    m[1] = { -cb * s, -sb * s };            // -e^{+i b} s
    m[2] = { cb * s, -sb * s };             // e^{-i b} s
    m[3] = { ca * c,  sa * c };             // e^{+i a} c
}

__device__ __forceinline__ void mat2mul(const C2 t[4], const C2 d[4], C2 m[4]) {
    m[0] = cmadd2(t[0], d[0], t[1], d[2]);
    m[1] = cmadd2(t[0], d[1], t[1], d[3]);
    m[2] = cmadd2(t[2], d[0], t[3], d[2]);
    m[3] = cmadd2(t[2], d[1], t[3], d[3]);
}

// Apply 2x2 gate on state bit position p (amp index s = lane*8 + r).
__device__ __forceinline__ void apply_gate(C2 st[8], const C2 m[4], int p, int lane) {
    if (p < 3) {
        // pairing inside the lane's 8 slots
        const int mask = 1 << p;
        #pragma unroll
        for (int r = 0; r < 8; ++r) {
            if ((r & mask) == 0) {
                const int r1 = r | mask;
                C2 a0 = st[r], a1 = st[r1];
                st[r]  = cmadd2(m[0], a0, m[1], a1);
                st[r1] = cmadd2(m[2], a0, m[3], a1);
            }
        }
    } else {
        // pairing across lanes: wave32 xor-shuffle
        const int lm  = 1 << (p - 3);
        const int bit = (lane >> (p - 3)) & 1;
        #pragma unroll
        for (int r = 0; r < 8; ++r) {
            C2 mine = st[r];
            C2 peer;
            peer.x = __shfl_xor(mine.x, lm, 32);
            peer.y = __shfl_xor(mine.y, lm, 32);
            C2 a0 = bit ? peer : mine;   // amplitude with bit p == 0
            C2 a1 = bit ? mine : peer;   // amplitude with bit p == 1
            C2 n0 = cmadd2(m[0], a0, m[1], a1);
            C2 n1 = cmadd2(m[2], a0, m[3], a1);
            st[r] = bit ? n1 : n0;
        }
    }
}

__global__ __launch_bounds__(THREADS)
void qcirc_wmma_kernel(const float* __restrict__ xg,
                       const float* __restrict__ theta,
                       const float* __restrict__ omega,
                       float* __restrict__ out, int B) {
    __shared__ _Float16 probs[ITEMS][LDSTR];

    const int lane      = threadIdx.x & 31;
    const int wave      = threadIdx.x >> 5;
    const int blockBase = blockIdx.x * ITEMS;
    int g = blockBase + wave;
    if (g >= B) g = B - 1;                    // clamp; store is guarded below

    // CDNA5 speculative prefetch of next block's feature rows (global_prefetch_b8)
    __builtin_prefetch(xg + (size_t)(blockBase + ITEMS) * 12, 0, 1);

    float x[12];
    #pragma unroll
    for (int i = 0; i < 12; ++i) x[i] = xg[g * 12 + i];

    // |0...0>  (s = lane*8 + r)
    C2 st[8];
    #pragma unroll
    for (int r = 0; r < 8; ++r) st[r] = { 0.f, 0.f };
    if (lane == 0) st[0].x = 1.f;

    #pragma unroll
    for (int l = 0; l < NL; ++l) {
        #pragma unroll
        for (int q = 0; q < NQ; ++q) {
            const int p = 7 - q;              // wire 0 = MSB
            const int f = (q & 3) * 3;        // feature block reused by vstacks
            C2 dm[4], tm[4], m[4];
            rot_mat(x[f + 0] * omega[l * 24 + q * 3 + 0],
                    x[f + 1] * omega[l * 24 + q * 3 + 1],
                    x[f + 2] * omega[l * 24 + q * 3 + 2], dm);
            rot_mat(theta[(q * 5 + l) * 3 + 0],
                    theta[(q * 5 + l) * 3 + 1],
                    theta[(q * 5 + l) * 3 + 2], tm);
            mat2mul(tm, dm, m);               // theta-Rot after data-Rot
            apply_gate(st, m, p, lane);
        }
        // CZ diagonal: even layers pairs (0,1)(2,3)(4,5)(6,7) -> bit-pair mask 0x55
        //              odd  layers pairs (1,2)(3,4)(5,6)      -> mask 0x2A
        const unsigned pm = (l & 1) ? 0x2Au : 0x55u;
        #pragma unroll
        for (int r = 0; r < 8; ++r) {
            unsigned s = (unsigned)(lane * 8 + r);
            unsigned t = s & (s >> 1);        // t_i = b_i & b_{i+1}
            if (__popc(t & pm) & 1) { st[r].x = -st[r].x; st[r].y = -st[r].y; }
        }
    }

    // final data-reuploading layer: single Rot with combined angles
    #pragma unroll
    for (int q = 0; q < NQ; ++q) {
        const int p = 7 - q;
        const int f = (q & 3) * 3;
        C2 m[4];
        rot_mat(x[f + 0] * omega[4 * 24 + q * 3 + 0] + theta[(q * 5 + 4) * 3 + 0],
                x[f + 1] * omega[4 * 24 + q * 3 + 1] + theta[(q * 5 + 4) * 3 + 1],
                x[f + 2] * omega[4 * 24 + q * 3 + 2] + theta[(q * 5 + 4) * 3 + 2], m);
        apply_gate(st, m, p, lane);
    }

    // probabilities -> LDS as f16 (WMMA A operand staging)
    #pragma unroll
    for (int r = 0; r < 8; ++r) {
        float pr = st[r].x * st[r].x + st[r].y * st[r].y;
        probs[wave][lane * 8 + r] = (_Float16)pr;
    }
    __syncthreads();

    // -----------------------------------------------------------------------
    // <Z_q> tile: D(16x16) = probs(16x256) @ Z(256x16), cols 8..15 zero-padded.
    // v_wmma_f32_16x16x32_f16, 8 K-steps, f32 accumulate. Wave 0 only
    // (EXEC all-ones inside this wave as WMMA requires).
    // -----------------------------------------------------------------------
    if (wave == 0) {
        const int row    = lane & 15;        // A row (item) / B column (qubit)
        const int hi     = lane >> 4;        // half-wave select
        const int kbaseA = hi * 16;          // A: contiguous k = 16*hi + s (this code's order)
        const int kbaseB = hi * 16;          // B: contiguous k = 16*hi + s, same order as A
        // Z[k, n] = ±1 -> f16 bit trick: 0x3C00 ^ (bit<<15); cols 8..15 zeroed
        const unsigned colmask = (row < 8) ? 0xFFFFFFFFu : 0u;
        const int      bitpos  = (7 - row) & 7;
        v8f acc = {};
        #pragma unroll
        for (int k0 = 0; k0 < DIM; k0 += 32) {
            v16h a;
            v8u  bw;
            #pragma unroll
            for (int j = 0; j < 8; ++j) {
                const int ka = k0 + kbaseA + 2 * j;
                a[2 * j]     = probs[row][ka];
                a[2 * j + 1] = probs[row][ka + 1];
                const int kb = k0 + kbaseB + 2 * j;
                const unsigned z0 = 0x3C00u ^ ((((unsigned)kb       >> bitpos) & 1u) << 15);
                const unsigned z1 = 0x3C00u ^ ((((unsigned)(kb + 1) >> bitpos) & 1u) << 15);
                bw[j] = (z0 | (z1 << 16)) & colmask;
            }
            const v16h b = __builtin_bit_cast(v16h, bw);
            acc = __builtin_amdgcn_wmma_f32_16x16x32_f16(
                /*neg_a=*/false, a, /*neg_b=*/false, b,
                /*c_mod=*/(short)0, acc, /*reuse_a=*/false, /*reuse_b=*/false);
        }
        // f32 D layout: lane n (0-15) holds col n, VGPR v = row v; lanes 16-31 rows v+8
        const int col = lane & 15;
        if (col < 8) {
            #pragma unroll
            for (int v = 0; v < 8; ++v) {
                const int mrow = v + hi * 8;
                const int item = blockBase + mrow;
                if (item < B) out[item * 8 + col] = acc[v];
            }
        }
    }
}

extern "C" void kernel_launch(void* const* d_in, const int* in_sizes, int n_in,
                              void* d_out, int out_size, void* d_ws, size_t ws_size,
                              hipStream_t stream) {
    const float* x     = (const float*)d_in[0];   // (B, 12) f32
    const float* theta = (const float*)d_in[1];   // (8, 5, 3) f32
    const float* omega = (const float*)d_in[2];   // (5, 24) f32
    float* out         = (float*)d_out;           // (B, 8) f32

    const int B      = in_sizes[0] / 12;
    const int blocks = (B + ITEMS - 1) / ITEMS;
    hipLaunchKernelGGL(qcirc_wmma_kernel, dim3(blocks), dim3(THREADS), 0, stream,
                       x, theta, omega, out, B);
}
